// RGCNRegression_13597866459798
// MI455X (gfx1250) — matrix operand order repaired and checked by reading, not compiled, not run on hardware
//
#include <hip/hip_runtime.h>
#include <hip/hip_bf16.h>

typedef __bf16 bf16x16 __attribute__((ext_vector_type(16)));
typedef float  f32x8   __attribute__((ext_vector_type(8)));

#define CDIM 128   // feature dim (C == H == 128)
#define RREL 8

// ---------------------------------------------------------------- zero
__global__ void rgcn_zero_kernel(float* __restrict__ p, long long n4) {
    long long i = (long long)blockIdx.x * blockDim.x + threadIdx.x;
    if (i < n4) {
        *(float4*)(p + i * 4) = make_float4(0.f, 0.f, 0.f, 0.f);
    }
}

// ---------------------------------------------------------------- edge counts
__global__ void rgcn_count_kernel(const int* __restrict__ etype,
                                  const int* __restrict__ dst,
                                  float* __restrict__ cnt, int E, int N) {
    int e = blockIdx.x * blockDim.x + threadIdx.x;
    if (e >= E) return;
    atomicAdd(cnt + (size_t)etype[e] * N + dst[e], 1.0f);
}

// cnt -> 1/max(cnt,1), in place
__global__ void rgcn_inv_kernel(float* __restrict__ cnt, int RN) {
    int t = blockIdx.x * blockDim.x + threadIdx.x;
    if (t >= RN) return;
    float c = cnt[t];
    cnt[t] = 1.0f / fmaxf(c, 1.0f);
}

// ---------------------------------------------------------------- weight pack
// Pack [root;W[0..7]] (both layers) into per-lane WMMA B fragments:
// segment s (0..8), kstep t (0..3), nblock j (0..7), lane l (0..31):
// 16 bf16 = B[k][n], n = j*16 + (l&15), k = t*32 + (l>>4)*8 + e + (e>=8 ? 8 : 0), e=0..15
__global__ void rgcn_pack_kernel(const float* __restrict__ root1,
                                 const float* __restrict__ W1,
                                 const float* __restrict__ root2,
                                 const float* __restrict__ W2,
                                 __bf16* __restrict__ bp1,
                                 __bf16* __restrict__ bp2) {
    int id = blockIdx.x * blockDim.x + threadIdx.x;
    if (id >= 2 * 9 * 4 * 8 * 32) return;
    int l = id & 31;
    int j = (id >> 5) & 7;
    int t = (id >> 8) & 3;
    int rest  = id >> 10;      // 0..17
    int s     = rest % 9;
    int layer = rest / 9;
    const float* Wseg = (layer == 0)
        ? ((s == 0) ? root1 : W1 + (size_t)(s - 1) * CDIM * CDIM)
        : ((s == 0) ? root2 : W2 + (size_t)(s - 1) * CDIM * CDIM);
    __bf16* out = ((layer == 0) ? bp1 : bp2) +
                  (size_t)((((s * 4 + t) * 8 + j) * 32) + l) * 16;
    int n     = j * 16 + (l & 15);
    int kbase = t * 32 + (l >> 4) * 8;
#pragma unroll
    for (int e = 0; e < 16; e++)
        out[e] = (__bf16)Wseg[(size_t)(kbase + e + (e >= 8 ? 8 : 0)) * CDIM + n];
}

// ---------------------------------------------------------------- edge scatter
// one lane per (edge, 4-float chunk): float4 gather of F[src], 4 f32 atomics
__global__ void rgcn_scatter_kernel(const float* __restrict__ F,
                                    const int* __restrict__ src,
                                    const int* __restrict__ dst,
                                    const int* __restrict__ etype,
                                    float* __restrict__ agg, int E, int N) {
    long long t = (long long)blockIdx.x * blockDim.x + threadIdx.x;
    int e = (int)(t >> 5);
    if (e >= E) return;
    int l = (int)(t & 31);
    int s = src[e], d = dst[e], r = etype[e];
    float4 v = *(const float4*)(F + (size_t)s * CDIM + l * 4);
    float* p = agg + ((size_t)r * N + d) * CDIM + l * 4;
    atomicAdd(p + 0, v.x);
    atomicAdd(p + 1, v.y);
    atomicAdd(p + 2, v.z);
    atomicAdd(p + 3, v.w);
}

// agg[rn][:] *= inv[rn]
__global__ void rgcn_norm_kernel(float* __restrict__ agg,
                                 const float* __restrict__ inv, int RN) {
    long long t = (long long)blockIdx.x * blockDim.x + threadIdx.x;
    int rn = (int)(t >> 5);
    if (rn >= RN) return;
    int l = (int)(t & 31);
    float s = inv[rn];
    float4* p = (float4*)(agg + (size_t)rn * CDIM + l * 4);
    float4 v = *p;
    v.x *= s; v.y *= s; v.z *= s; v.w *= s;
    *p = v;
}

// build a 16-wide bf16 A fragment (rows = lane&15, K = khalf*8 + {0..7,16..23})
__device__ __forceinline__ bf16x16 rgcn_load_a_frag(const float* __restrict__ ap) {
    float4 p0 = *(const float4*)(ap + 0);
    float4 p1 = *(const float4*)(ap + 4);
    float4 q0 = *(const float4*)(ap + 16);
    float4 q1 = *(const float4*)(ap + 20);
    bf16x16 a;
    a[0]  = (__bf16)p0.x; a[1]  = (__bf16)p0.y;
    a[2]  = (__bf16)p0.z; a[3]  = (__bf16)p0.w;
    a[4]  = (__bf16)p1.x; a[5]  = (__bf16)p1.y;
    a[6]  = (__bf16)p1.z; a[7]  = (__bf16)p1.w;
    a[8]  = (__bf16)q0.x; a[9]  = (__bf16)q0.y;
    a[10] = (__bf16)q0.z; a[11] = (__bf16)q0.w;
    a[12] = (__bf16)q1.x; a[13] = (__bf16)q1.y;
    a[14] = (__bf16)q1.z; a[15] = (__bf16)q1.w;
    return a;
}

// ---------------------------------------------------------------- WMMA GEMM
// out[N x 128] = ReLU(bias + F@root + sum_r agg[r]@W[r])
// block = 8 waves; wave computes a 32-row x 128-col stripe (2 A frags,
// 16 accumulators); B fragments are software-pipelined and each feeds
// two back-to-back WMMAs.
__global__ __launch_bounds__(256)
void rgcn_gemm_kernel(const float* __restrict__ F,
                      const float* __restrict__ agg,
                      const __bf16* __restrict__ Bpack,
                      const float* __restrict__ bias,
                      float* __restrict__ out, int N) {
    const int lane  = threadIdx.x & 31;
    const int wave  = threadIdx.x >> 5;
    const int m0    = (blockIdx.x * 8 + wave) * 32;
    if (m0 >= N) return;
    const int row   = lane & 15;
    const int khalf = lane >> 4;

    f32x8 acc0[8], acc1[8];
#pragma unroll
    for (int j = 0; j < 8; j++)
#pragma unroll
        for (int v = 0; v < 8; v++) { acc0[j][v] = 0.0f; acc1[j][v] = 0.0f; }

    const int r0 = m0 + row;                 // always < N (N % 16 == 0)
    int r1 = m0 + 16 + row;                  // may run off the end on last tile
    if (r1 >= N) r1 = N - 1;

    for (int s = 0; s < 9; s++) {
        const float* Abase = (s == 0) ? F : (agg + (size_t)(s - 1) * (size_t)N * CDIM);
        const float* a0p = Abase + (size_t)r0 * CDIM + khalf * 8;
        const float* a1p = Abase + (size_t)r1 * CDIM + khalf * 8;
        const __bf16* bseg = Bpack + (size_t)s * (4 * 8 * 32 * 16);
#pragma unroll
        for (int t = 0; t < 4; t++) {
            bf16x16 a0 = rgcn_load_a_frag(a0p + t * 32);
            bf16x16 a1 = rgcn_load_a_frag(a1p + t * 32);
            const __bf16* bt = bseg + (size_t)t * (8 * 32 * 16) + (size_t)lane * 16;
            bf16x16 bcur = *(const bf16x16*)(bt);
#pragma unroll
            for (int j = 0; j < 8; j++) {
                bf16x16 bnext = (j < 7)
                    ? *(const bf16x16*)(bt + (size_t)(j + 1) * 32 * 16)
                    : bcur;
                acc0[j] = __builtin_amdgcn_wmma_f32_16x16x32_bf16(
                    false, a0, false, bcur, (short)0, acc0[j], false, false);
                acc1[j] = __builtin_amdgcn_wmma_f32_16x16x32_bf16(
                    false, a1, false, bcur, (short)0, acc1[j], false, false);
                bcur = bnext;
            }
        }
    }

    const int nlo = lane & 15;
    float bv[8];
#pragma unroll
    for (int j = 0; j < 8; j++) bv[j] = bias[j * 16 + nlo];
    const int rb0 = m0 + khalf * 8;
    const int rb1 = m0 + 16 + khalf * 8;
#pragma unroll
    for (int j = 0; j < 8; j++) {
#pragma unroll
        for (int v = 0; v < 8; v++) {
            float val = acc0[j][v] + bv[j];
            val = val > 0.0f ? val : 0.0f;
            out[(size_t)(rb0 + v) * CDIM + j * 16 + nlo] = val;
            if (rb1 + v < N) {
                float val1 = acc1[j][v] + bv[j];
                val1 = val1 > 0.0f ? val1 : 0.0f;
                out[(size_t)(rb1 + v) * CDIM + j * 16 + nlo] = val1;
            }
        }
    }
}

// ---------------------------------------------------------------- final head
__global__ void rgcn_out_kernel(const float* __restrict__ h,
                                const float* __restrict__ Wout,
                                const float* __restrict__ bout,
                                float* __restrict__ out, int N) {
    long long t = (long long)blockIdx.x * blockDim.x + threadIdx.x;
    int n = (int)(t >> 5);
    if (n >= N) return;
    int l = (int)(t & 31);
    float4 a = *(const float4*)(h + (size_t)n * CDIM + l * 4);
    float4 w = *(const float4*)(Wout + l * 4);
    float sum = a.x * w.x + a.y * w.y + a.z * w.z + a.w * w.w;
#pragma unroll
    for (int off = 16; off > 0; off >>= 1) sum += __shfl_xor(sum, off, 32);
    if (l == 0) out[n] = sum + bout[0];
}

// ---------------------------------------------------------------- launch
extern "C" void kernel_launch(void* const* d_in, const int* in_sizes, int n_in,
                              void* d_out, int out_size, void* d_ws, size_t ws_size,
                              hipStream_t stream) {
    const float* x     = (const float*)d_in[0];
    const int*   eidx  = (const int*)d_in[1];
    const int*   etype = (const int*)d_in[2];
    const float* W1    = (const float*)d_in[3];
    const float* root1 = (const float*)d_in[4];
    const float* b1    = (const float*)d_in[5];
    const float* W2    = (const float*)d_in[6];
    const float* root2 = (const float*)d_in[7];
    const float* b2    = (const float*)d_in[8];
    const float* Wout  = (const float*)d_in[9];
    const float* bout  = (const float*)d_in[10];
    float* out = (float*)d_out;

    const int N = in_sizes[0] / CDIM;
    const int E = in_sizes[2];
    const int R = RREL;
    const int RN = R * N;
    const int* src = eidx;
    const int* dst = eidx + E;

    // workspace carve-out (256B aligned)
    size_t off = 0;
    auto carve = [&](size_t bytes) {
        size_t o = off;
        off = (off + bytes + 255) & ~(size_t)255;
        return o;
    };
    char* ws = (char*)d_ws;
    float*  agg = (float*)(ws + carve((size_t)RN * CDIM * sizeof(float)));
    float*  cnt = (float*)(ws + carve((size_t)RN * sizeof(float)));
    float*  h1  = (float*)(ws + carve((size_t)N * CDIM * sizeof(float)));
    float*  h2  = (float*)(ws + carve((size_t)N * CDIM * sizeof(float)));
    __bf16* bp1 = (__bf16*)(ws + carve((size_t)9 * 4 * 8 * 32 * 16 * sizeof(__bf16)));
    __bf16* bp2 = (__bf16*)(ws + carve((size_t)9 * 4 * 8 * 32 * 16 * sizeof(__bf16)));
    (void)ws_size; (void)n_in; (void)out_size;

    const int TB = 256;
    const long long agg_f4 = (long long)RN * CDIM / 4;
    const long long cnt_f4 = (RN + 3) / 4;

    // counts (shared by both layers) + packed weights
    rgcn_zero_kernel<<<(unsigned)((cnt_f4 + TB - 1) / TB), TB, 0, stream>>>(cnt, cnt_f4);
    rgcn_count_kernel<<<(E + TB - 1) / TB, TB, 0, stream>>>(etype, dst, cnt, E, N);
    rgcn_inv_kernel<<<(RN + TB - 1) / TB, TB, 0, stream>>>(cnt, RN);
    rgcn_pack_kernel<<<(2 * 9 * 4 * 8 * 32 + TB - 1) / TB, TB, 0, stream>>>(
        root1, W1, root2, W2, bp1, bp2);

    const long long scat_th = (long long)E * 32;
    const long long norm_th = (long long)RN * 32;
    const int gemm_blocks = (((N + 31) / 32) + 7) / 8;

    // ---- layer 1: x -> h1
    rgcn_zero_kernel<<<(unsigned)((agg_f4 + TB - 1) / TB), TB, 0, stream>>>(agg, agg_f4);
    rgcn_scatter_kernel<<<(unsigned)((scat_th + TB - 1) / TB), TB, 0, stream>>>(
        x, src, dst, etype, agg, E, N);
    rgcn_norm_kernel<<<(unsigned)((norm_th + TB - 1) / TB), TB, 0, stream>>>(agg, cnt, RN);
    rgcn_gemm_kernel<<<gemm_blocks, TB, 0, stream>>>(x, agg, bp1, b1, h1, N);

    // ---- layer 2: h1 -> h2
    rgcn_zero_kernel<<<(unsigned)((agg_f4 + TB - 1) / TB), TB, 0, stream>>>(agg, agg_f4);
    rgcn_scatter_kernel<<<(unsigned)((scat_th + TB - 1) / TB), TB, 0, stream>>>(
        h1, src, dst, etype, agg, E, N);
    rgcn_norm_kernel<<<(unsigned)((norm_th + TB - 1) / TB), TB, 0, stream>>>(agg, cnt, RN);
    rgcn_gemm_kernel<<<gemm_blocks, TB, 0, stream>>>(h1, agg, bp2, b2, h2, N);

    // ---- head
    const long long out_th = (long long)N * 32;
    rgcn_out_kernel<<<(unsigned)((out_th + TB - 1) / TB), TB, 0, stream>>>(
        h2, Wout, bout, out, N);
}
